// GraphTransformer_81286551044269
// MI455X (gfx1250) — matrix operand order repaired and checked by reading, not compiled, not run on hardware
//
#include <hip/hip_runtime.h>

// -------------------- problem constants (match reference) --------------------
#define N_NODES 50000
#define N_EDGES 800000
#define IN_DIM  128
#define HID     32
#define HEADS   4
#define C1      (HEADS * HID)   // 128, layer-1 channel count
#define C2      (HID)           // 32,  layer-2 channel count

// -------------------- CDNA5 WMMA types --------------------
typedef __bf16 bf16_t;
typedef __attribute__((ext_vector_type(16))) __bf16 v16bf;
typedef __attribute__((ext_vector_type(8)))  __bf16 v8bf;
typedef __attribute__((ext_vector_type(8)))  float  v8f;

// -------------------- workspace layout (bytes) --------------------
// all offsets multiple of 256 (hipMalloc base is 256-aligned)
#define OFF_XB  ((size_t)0)            // bf16  N*128            = 12,800,000 B
#define OFF_WP  ((size_t)12800000)     // bf16 packed weights    =    163,840 B
#define OFF_Q   ((size_t)12963840)     // f32   N*128
#define OFF_K   ((size_t)38563840)     // f32   N*128
#define OFF_V   ((size_t)64163840)     // f32   N*128
#define OFF_H   ((size_t)89763840)     // f32   N*128 (layer1 h; later layer2 h2 N*32)
#define OFF_LG  ((size_t)115363840)    // f32   E*4   (edge logits / p)
#define OFF_M   ((size_t)128163840)    // f32   N*4   (segment max)
#define OFF_D   ((size_t)128963840)    // f32   N*4   (segment denom)
#define OFF_G   ((size_t)129763840)    // f32   32    (pooled column sums)

// packed-weight sub-offsets (in bf16 elements)
#define WP_Q1 0
#define WP_K1 16384
#define WP_V1 32768
#define WP_S1 49152
#define WP_Q2 65536
#define WP_K2 69632
#define WP_V2 73728
#define WP_S2 77824

// -------------------- helpers --------------------
__device__ __forceinline__ void atomicMaxF(float* addr, float val) {
    int* ia = (int*)addr;
    int old = __float_as_int(*addr);
    while (__int_as_float(old) < val) {
        int assumed = old;
        old = atomicCAS(ia, assumed, __float_as_int(val));
        if (old == assumed) break;
    }
}

// -------------------- weight packing into WMMA B-fragment order --------------
// W: [K x OUTN] row-major f32. Packed layout (bf16):
//   wp[(((tn*KT + tk)*32 + lane)*16 + i] = W[(tk*32 + (lane<16?0:8) + (i<8?i:i+8))*OUTN + tn*16 + (lane&15)]
// so each lane's 16 halves for a k-tile are one contiguous 32-byte chunk.
__global__ void k_pack_w(const float* __restrict__ W, bf16_t* __restrict__ Wp,
                         int K, int OUTN) {
    int tid = blockIdx.x * blockDim.x + threadIdx.x;
    int total = K * OUTN;
    if (tid >= total) return;
    int i    = tid & 15;
    int lane = (tid >> 4) & 31;
    int tile = tid >> 9;
    int KT   = K >> 5;
    int tk   = tile % KT;
    int tn   = tile / KT;
    int n    = tn * 16 + (lane & 15);
    int kk   = tk * 32 + ((lane < 16) ? 0 : 8) + ((i < 8) ? i : i + 8);
    Wp[tid] = (bf16_t)W[kk * OUTN + n];
}

// -------------------- f32 -> bf16 convert (optional relu) --------------------
__global__ void k_cvt_bf16(const float* __restrict__ X, bf16_t* __restrict__ Y,
                           int n, int relu) {
    int i = blockIdx.x * blockDim.x + threadIdx.x;
    if (i >= n) return;
    float v = X[i];
    if (relu && v < 0.f) v = 0.f;
    Y[i] = (bf16_t)v;
}

// -------------------- init segment-max / denom / pool accumulators -----------
__global__ void k_init_md(float* __restrict__ m, float* __restrict__ dsum,
                          float* __restrict__ g, int n) {
    int i = blockIdx.x * blockDim.x + threadIdx.x;
    if (i < n) {
        m[i]    = __int_as_float(0xff800000); // -inf
        dsum[i] = 0.f;
    }
    if (g != nullptr && i < HID) g[i] = 0.f;
}

// -------------------- WMMA GEMM: Y[M,OUTN] = bf16(X)[M,K] @ W + bias ---------
// One wave computes a 16 x (16*NB) output tile: the A fragment is loaded once
// per k-step and reused across NB column WMMAs (register blocking), giving
// 2 + 2*NB b128-loads per NB wmma (2.5 loads/wmma at NB=4) instead of 4/wmma.
// K multiple of 32; M multiple of 16 (50000 = 3125*16) so EXEC is all-ones.
// A 16x32 bf16 layout (per ISA table): lane<16 row=lane, halves 0..7 -> K 0..7,
//   halves 8..15 -> K 16..23 ; lane>=16 row=lane-16, K 8..15 and 24..31.
template <int NB>
__global__ __launch_bounds__(32)
void k_gemm_wmma(const bf16_t* __restrict__ X, const bf16_t* __restrict__ Wp,
                 const float* __restrict__ bias, float* __restrict__ Y,
                 int K, int OUTN) {
    const int lane = threadIdx.x;            // 0..31
    const int tm   = blockIdx.x;             // row tile
    const int tn0  = blockIdx.y * NB;        // first 16-wide column tile
    const int KT   = K >> 5;

    v8f c[NB];
#pragma unroll
    for (int b = 0; b < NB; ++b) c[b] = (v8f){0.f, 0.f, 0.f, 0.f, 0.f, 0.f, 0.f, 0.f};

    const int row = tm * 16 + (lane & 15);
    const int kA  = (lane < 16) ? 0 : 8;
    const bf16_t* xrow = X + (size_t)row * K;
    // per-column-tile packed-B bases (each lane's chunk is 32B contiguous)
    const bf16_t* wbase[NB];
#pragma unroll
    for (int b = 0; b < NB; ++b)
        wbase[b] = Wp + ((size_t)((tn0 + b) * KT) * 32 + lane) * 16;

#pragma unroll 4
    for (int kt = 0; kt < KT; ++kt) {
        const int k0 = kt * 32;
        v8bf alo = *(const v8bf*)(xrow + k0 + kA);
        v8bf ahi = *(const v8bf*)(xrow + k0 + kA + 16);
        v16bf a = __builtin_shufflevector(alo, ahi,
                                          0,1,2,3,4,5,6,7,8,9,10,11,12,13,14,15);
#pragma unroll
        for (int b = 0; b < NB; ++b) {
            v16bf bb = *(const v16bf*)(wbase[b] + (size_t)kt * (32 * 16));
            c[b] = __builtin_amdgcn_wmma_f32_16x16x32_bf16(
                       /*neg_a=*/false, a, /*neg_b=*/false, bb,
                       /*c_mod=*/(short)0, c[b],
                       /*reuse_a=*/false, /*reuse_b=*/false);
        }
    }

    // C/D layout: VGPR j -> row (tm*16 + j + (lane<16?0:8)), col tile*16 + (lane&15)
    const int rbase = tm * 16 + ((lane < 16) ? 0 : 8);
#pragma unroll
    for (int b = 0; b < NB; ++b) {
        const int col  = (tn0 + b) * 16 + (lane & 15);
        const float bc = bias[col];
#pragma unroll
        for (int j = 0; j < 8; ++j) {
            Y[(size_t)(rbase + j) * OUTN + col] = c[b][j] + bc;
        }
    }
}

// -------------------- per-edge attention logits + segment max ----------------
__global__ void k_logits(const float* __restrict__ q, const float* __restrict__ k,
                         const int* __restrict__ ei, float* __restrict__ logits,
                         float* __restrict__ m, int E, int NH, int D) {
    int idx = blockIdx.x * blockDim.x + threadIdx.x;
    if (idx >= E * NH) return;
    int e = idx / NH, h = idx - e * NH;
    int src = ei[e], dst = ei[E + e];
    const float* qr = q + (size_t)dst * (NH * D) + h * D;
    const float* kr = k + (size_t)src * (NH * D) + h * D;
    float acc = 0.f;
#pragma unroll 8
    for (int j = 0; j < D; ++j) acc += qr[j] * kr[j];
    acc *= rsqrtf((float)D);
    logits[idx] = acc;
    atomicMaxF(&m[dst * NH + h], acc);
}

// -------------------- p = exp(logit - max); denom accumulation ---------------
__global__ void k_softp(const int* __restrict__ ei, float* __restrict__ logits,
                        const float* __restrict__ m, float* __restrict__ dsum,
                        int E, int NH) {
    int idx = blockIdx.x * blockDim.x + threadIdx.x;
    if (idx >= E * NH) return;
    int e = idx / NH, h = idx - e * NH;
    int dst = ei[E + e];
    float p = __expf(logits[idx] - m[dst * NH + h]);
    logits[idx] = p;
    atomicAdd(&dsum[dst * NH + h], p);
}

// -------------------- scatter alpha * v[src] into Y[dst] ---------------------
__global__ void k_scatter(const int* __restrict__ ei, const float* __restrict__ logits,
                          const float* __restrict__ dsum, const float* __restrict__ V,
                          float* __restrict__ Y, int E, int NH, int D) {
    const int C = NH * D;
    int idx = blockIdx.x * blockDim.x + threadIdx.x;
    if (idx >= E * C) return;
    int e = idx / C, c = idx - e * C;
    int h = c / D;
    int src = ei[e], dst = ei[E + e];
    float alpha = logits[(size_t)e * NH + h] / dsum[dst * NH + h];
    atomicAdd(&Y[(size_t)dst * C + c], alpha * V[(size_t)src * C + c]);
}

// -------------------- relu + column-sum pooling ------------------------------
__global__ void k_pool(const float* __restrict__ H, float* __restrict__ g) {
    int tid = blockIdx.x * blockDim.x + threadIdx.x;
    int c = tid & 31;
    int row = tid >> 5;
    int rstride = (gridDim.x * blockDim.x) >> 5;
    float acc = 0.f;
    for (; row < N_NODES; row += rstride) {
        float v = H[(size_t)row * HID + c];
        acc += (v > 0.f) ? v : 0.f;
    }
    atomicAdd(&g[c], acc);
}

// -------------------- final: out = (g/N) @ Wo + bo ---------------------------
__global__ __launch_bounds__(32)
void k_final(const float* __restrict__ g, const float* __restrict__ Wo,
             const float* __restrict__ bo, float* __restrict__ out, float invN) {
    int lane = threadIdx.x;
    float v = g[lane] * invN * Wo[lane];
#pragma unroll
    for (int off = 16; off > 0; off >>= 1) v += __shfl_down(v, off, 32);
    if (lane == 0) out[0] = v + bo[0];
}

// ============================================================================
extern "C" void kernel_launch(void* const* d_in, const int* in_sizes, int n_in,
                              void* d_out, int out_size, void* d_ws, size_t ws_size,
                              hipStream_t stream) {
    (void)in_sizes; (void)n_in; (void)out_size; (void)ws_size;

    const float* x   = (const float*)d_in[0];
    const int*   ei  = (const int*)d_in[1];
    const float* Wq1 = (const float*)d_in[2];  const float* bq1 = (const float*)d_in[3];
    const float* Wk1 = (const float*)d_in[4];  const float* bk1 = (const float*)d_in[5];
    const float* Wv1 = (const float*)d_in[6];  const float* bv1 = (const float*)d_in[7];
    const float* Ws1 = (const float*)d_in[8];  const float* bs1 = (const float*)d_in[9];
    const float* Wq2 = (const float*)d_in[10]; const float* bq2 = (const float*)d_in[11];
    const float* Wk2 = (const float*)d_in[12]; const float* bk2 = (const float*)d_in[13];
    const float* Wv2 = (const float*)d_in[14]; const float* bv2 = (const float*)d_in[15];
    const float* Ws2 = (const float*)d_in[16]; const float* bs2 = (const float*)d_in[17];
    const float* Wo  = (const float*)d_in[18]; const float* bo  = (const float*)d_in[19];

    char* ws = (char*)d_ws;
    bf16_t* xb  = (bf16_t*)(ws + OFF_XB);
    bf16_t* wp  = (bf16_t*)(ws + OFF_WP);
    float* qb   = (float*)(ws + OFF_Q);
    float* kb   = (float*)(ws + OFF_K);
    float* vb   = (float*)(ws + OFF_V);
    float* hb   = (float*)(ws + OFF_H);
    float* lg   = (float*)(ws + OFF_LG);
    float* mb   = (float*)(ws + OFF_M);
    float* db   = (float*)(ws + OFF_D);
    float* gb   = (float*)(ws + OFF_G);

    // ---- pack all weights into WMMA B-fragment order (bf16) ----
    k_pack_w<<<(IN_DIM * C1 + 255) / 256, 256, 0, stream>>>(Wq1, wp + WP_Q1, IN_DIM, C1);
    k_pack_w<<<(IN_DIM * C1 + 255) / 256, 256, 0, stream>>>(Wk1, wp + WP_K1, IN_DIM, C1);
    k_pack_w<<<(IN_DIM * C1 + 255) / 256, 256, 0, stream>>>(Wv1, wp + WP_V1, IN_DIM, C1);
    k_pack_w<<<(IN_DIM * C1 + 255) / 256, 256, 0, stream>>>(Ws1, wp + WP_S1, IN_DIM, C1);
    k_pack_w<<<(C1 * C2 + 255) / 256, 256, 0, stream>>>(Wq2, wp + WP_Q2, C1, C2);
    k_pack_w<<<(C1 * C2 + 255) / 256, 256, 0, stream>>>(Wk2, wp + WP_K2, C1, C2);
    k_pack_w<<<(C1 * C2 + 255) / 256, 256, 0, stream>>>(Wv2, wp + WP_V2, C1, C2);
    k_pack_w<<<(C1 * C2 + 255) / 256, 256, 0, stream>>>(Ws2, wp + WP_S2, C1, C2);

    // ---- layer 1 ----
    k_cvt_bf16<<<(N_NODES * IN_DIM + 255) / 256, 256, 0, stream>>>(x, xb, N_NODES * IN_DIM, 0);
    k_init_md<<<(N_NODES * HEADS + 255) / 256, 256, 0, stream>>>(mb, db, gb, N_NODES * HEADS);

    // 16x64 tile per wave: grid (3125, 128/64)
    dim3 g1(N_NODES / 16, C1 / (16 * 4));
    k_gemm_wmma<4><<<g1, 32, 0, stream>>>(xb, wp + WP_Q1, bq1, qb, IN_DIM, C1);
    k_gemm_wmma<4><<<g1, 32, 0, stream>>>(xb, wp + WP_K1, bk1, kb, IN_DIM, C1);
    k_gemm_wmma<4><<<g1, 32, 0, stream>>>(xb, wp + WP_V1, bv1, vb, IN_DIM, C1);
    k_gemm_wmma<4><<<g1, 32, 0, stream>>>(xb, wp + WP_S1, bs1, hb, IN_DIM, C1); // skip/root init

    k_logits <<<(N_EDGES * HEADS + 255) / 256, 256, 0, stream>>>(qb, kb, ei, lg, mb, N_EDGES, HEADS, HID);
    k_softp  <<<(N_EDGES * HEADS + 255) / 256, 256, 0, stream>>>(ei, lg, mb, db, N_EDGES, HEADS);
    k_scatter<<<(N_EDGES * C1 + 255) / 256, 256, 0, stream>>>(ei, lg, db, vb, hb, N_EDGES, HEADS, HID);

    // ---- relu + requantize h -> bf16 input for layer 2 ----
    k_cvt_bf16<<<(N_NODES * C1 + 255) / 256, 256, 0, stream>>>(hb, xb, N_NODES * C1, 1);

    // ---- layer 2 (heads=1, d=32) ----
    k_init_md<<<(N_NODES + 255) / 256, 256, 0, stream>>>(mb, db, nullptr, N_NODES);

    // 16x32 tile per wave: grid (3125, 1)
    dim3 g2(N_NODES / 16, C2 / (16 * 2));
    k_gemm_wmma<2><<<g2, 32, 0, stream>>>(xb, wp + WP_Q2, bq2, qb, C1, C2);
    k_gemm_wmma<2><<<g2, 32, 0, stream>>>(xb, wp + WP_K2, bk2, kb, C1, C2);
    k_gemm_wmma<2><<<g2, 32, 0, stream>>>(xb, wp + WP_V2, bv2, vb, C1, C2);
    k_gemm_wmma<2><<<g2, 32, 0, stream>>>(xb, wp + WP_S2, bs2, hb, C1, C2);   // h2 skip init

    k_logits <<<(N_EDGES + 255) / 256, 256, 0, stream>>>(qb, kb, ei, lg, mb, N_EDGES, 1, HID);
    k_softp  <<<(N_EDGES + 255) / 256, 256, 0, stream>>>(ei, lg, mb, db, N_EDGES, 1);
    k_scatter<<<(N_EDGES * C2 + 255) / 256, 256, 0, stream>>>(ei, lg, db, vb, hb, N_EDGES, 1, HID);

    // ---- relu + mean pool + output linear ----
    k_pool<<<64, 256, 0, stream>>>(hb, gb);
    k_final<<<1, 32, 0, stream>>>(gb, Wo, bo, (float*)d_out, 1.0f / (float)N_NODES);
}
